// WindowClassAttention_48524540510952
// MI455X (gfx1250) — matrix operand order repaired and checked by reading, not compiled, not run on hardware
//
#include <hip/hip_runtime.h>
#include <hip/hip_bf16.h>

typedef __bf16 bf16;
typedef __attribute__((ext_vector_type(16))) __bf16 v16bf;
typedef __attribute__((ext_vector_type(8)))  __bf16 v8bf;
typedef __attribute__((ext_vector_type(8)))  float  v8f;

#define B_WIN   1024
#define N_TOK   49
#define N_PAD   64
#define C_DIM   256
#define H_NUM   8
#define HD      32
#define CT_DIM  64
#define CTH     8
#define TC_DIM  384
#define TH_DIM  48
#define NW      64
#define BH_NUM  (B_WIN * H_NUM)       // 8192
#define ROWS    (B_WIN * N_TOK)       // 50176
#define RT      (ROWS / 16)           // 3136 row tiles
#define SCALE   0.17677669529663687f  // 32^-0.5

// (neg_a, A, neg_b, B, c_mod, C, reuse_a, reuse_b) — reuse flags are literals.
#define WMMA(a, b, c, ra) \
  __builtin_amdgcn_wmma_f32_16x16x32_bf16(false, (a), false, (b), (short)0, (c), (ra), false)

static __device__ __forceinline__ v16bf cat16(v8bf lo, v8bf hi) {
  return __builtin_shufflevector(lo, hi, 0,1,2,3,4,5,6,7,8,9,10,11,12,13,14,15);
}

// A fragment from bf16 row-major source with 16B-aligned rows.
// A k-order used here (same as packed B): lane<16 -> K {k0..k0+15}; lane>=16 -> K {k0+16..k0+31}.
static __device__ __forceinline__ v16bf load_a_bf16(const bf16* __restrict__ A,
                                                    int lda, int row, int k0, int lh) {
  const bf16* p = A + row * lda + k0 + lh * 16;
  v8bf lo = *(const v8bf*)p;
  v8bf hi = *(const v8bf*)(p + 8);
  return cat16(lo, hi);
}

// B fragment (contiguous, pre-packed): two aligned 16B loads per lane.
static __device__ __forceinline__ v16bf load_b_packed(const bf16* __restrict__ Wp) {
  v8bf lo = *(const v8bf*)Wp;
  v8bf hi = *(const v8bf*)(Wp + 8);
  return cat16(lo, hi);
}

// Cooperative async stage of a 16 x KDIM bf16 tile (rows contiguous) into LDS.
// gfx1250 GLOBAL_LOAD_ASYNC_TO_LDS_B128 (ASYNCcnt) + s_wait_asynccnt.
template<int KDIM>
static __device__ __forceinline__ void stage_a(const bf16* __restrict__ A, long m0,
                                               bf16* lds) {
  const unsigned long long gbase = (unsigned long long)(const void*)(A + m0 * KDIM);
  const unsigned lbase = (unsigned)(unsigned long long)(void*)lds;
  constexpr int chunks = (16 * KDIM * 2) / (128 * 16);
  #pragma unroll
  for (int c = 0; c < chunks; ++c) {
    const unsigned off = (unsigned)((c * 128 + threadIdx.x) * 16);
    asm volatile("global_load_async_to_lds_b128 %0, %1, %2"
                 :: "v"(lbase + off), "v"(off), "s"(gbase)
                 : "memory");
  }
  asm volatile("s_wait_asynccnt 0" ::: "memory");
  __syncthreads();
}

// 4 column tiles per wave: A fragment loaded once per K-step, reused 4x (RA hint).
template<int KDIM, int NT>
static __device__ __forceinline__ void mm_core4(const bf16* lds,
                                                const bf16* __restrict__ Wp,
                                                int nt0, int lane, v8f acc[4]) {
  const int lh = lane >> 4, ll = lane & 15;
  #pragma unroll
  for (int kt = 0; kt < KDIM / 32; ++kt) {
    v16bf a = load_a_bf16(lds, KDIM, ll, kt * 32, lh);
    const bf16* bp = Wp + ((kt * NT + nt0) * 32 + lane) * 16;
    acc[0] = WMMA(a, load_b_packed(bp),        acc[0], true);
    acc[1] = WMMA(a, load_b_packed(bp + 512),  acc[1], true);
    acc[2] = WMMA(a, load_b_packed(bp + 1024), acc[2], true);
    acc[3] = WMMA(a, load_b_packed(bp + 1536), acc[3], false);
  }
}

template<int KDIM, int NT>
static __device__ __forceinline__ void mm_core2(const bf16* lds,
                                                const bf16* __restrict__ Wp,
                                                int nt0, int lane, v8f acc[2]) {
  const int lh = lane >> 4, ll = lane & 15;
  #pragma unroll
  for (int kt = 0; kt < KDIM / 32; ++kt) {
    v16bf a = load_a_bf16(lds, KDIM, ll, kt * 32, lh);
    const bf16* bp = Wp + ((kt * NT + nt0) * 32 + lane) * 16;
    acc[0] = WMMA(a, load_b_packed(bp),       acc[0], true);
    acc[1] = WMMA(a, load_b_packed(bp + 512), acc[1], false);
  }
}

template<int KDIM, int NT>
static __device__ __forceinline__ v8f mm_core1(const bf16* lds,
                                               const bf16* __restrict__ Wp,
                                               int nt, int lane) {
  const int lh = lane >> 4, ll = lane & 15;
  v8f acc = {};
  #pragma unroll
  for (int kt = 0; kt < KDIM / 32; ++kt) {
    v16bf a = load_a_bf16(lds, KDIM, ll, kt * 32, lh);
    const bf16* bp = Wp + ((kt * NT + nt) * 32 + lane) * 16;
    acc = WMMA(a, load_b_packed(bp), acc, false);
  }
  return acc;
}

static __device__ __forceinline__ void win_decomp(int m0, int& bi0, int& nr0) {
  bi0 = m0 / N_TOK;
  nr0 = m0 - bi0 * N_TOK;
}

// ---------------- fp32 -> bf16 bulk convert -----------------------------------------
__global__ __launch_bounds__(256)
void k_cvt_bf16(const float* __restrict__ in, bf16* __restrict__ out, int n) {
  int i = (blockIdx.x * 256 + threadIdx.x) * 4;
  if (i >= n) return;
  float4 v = *(const float4*)(in + i);
  out[i + 0] = (bf16)v.x; out[i + 1] = (bf16)v.y;
  out[i + 2] = (bf16)v.z; out[i + 3] = (bf16)v.w;
}

// ---------------- pack fp32 weight (K x Nc) into B-fragment lane layout -------------
__global__ __launch_bounds__(256)
void k_pack_w(const float* __restrict__ W, int Kd, int Nc, bf16* __restrict__ out) {
  int i = blockIdx.x * 256 + threadIdx.x;
  if (i >= Kd * Nc) return;
  int e    = i & 15;
  int lane = (i >> 4) & 31;
  int tile = i >> 9;
  int NT = Nc >> 4;
  int nt = tile % NT;
  int kt = tile / NT;
  int k   = kt * 32 + (lane >> 4) * 16 + e;
  int col = nt * 16 + (lane & 15);
  out[i] = (bf16)W[k * Nc + col];
}

// ---------------- qkv GEMM: q/k n-major, v transposed d-major -----------------------
__global__ __launch_bounds__(128)
void k_qkv(const bf16* __restrict__ xb, const bf16* __restrict__ Wp,
           const float* __restrict__ bias, bf16* __restrict__ qb,
           bf16* __restrict__ kb, bf16* __restrict__ vbT) {
  __shared__ __attribute__((aligned(16))) bf16 At[16 * C_DIM];
  const int lane = threadIdx.x & 31, wave = threadIdx.x >> 5;
  const int lh = lane >> 4, ll = lane & 15;
  const int m0 = blockIdx.x * 16;
  const int nt0 = (blockIdx.y * 4 + wave) * 4;
  stage_a<C_DIM>(xb, m0, At);
  v8f acc[4] = {{}, {}, {}, {}};
  mm_core4<C_DIM, 48>(At, Wp, nt0, lane, acc);

  int bi0, nr0; win_decomp(m0, bi0, nr0);
  #pragma unroll
  for (int j = 0; j < 4; ++j) {
    const int col = (nt0 + j) * 16 + ll;
    const int which = col >> 8;         // 0=q 1=k 2=v (uniform per wave+j)
    const int c = col & 255;
    const int h = c >> 5, d = c & 31;
    const float bv = bias[col];
    const float scl = (which == 0) ? SCALE : 1.0f;
    #pragma unroll
    for (int r = 0; r < 8; ++r) {
      int t = nr0 + r + lh * 8;
      int bi = bi0 + (t >= N_TOK);
      int n  = t - ((t >= N_TOK) ? N_TOK : 0);
      float v = (acc[j][r] + bv) * scl;
      int bh = bi * H_NUM + h;
      if (which == 2)
        vbT[(bh * HD + d) * N_PAD + n] = (bf16)v;                    // (B,H,hd,Npad)
      else
        ((which == 0) ? qb : kb)[(bh * N_PAD + n) * HD + d] = (bf16)v;
    }
  }
}

// ---------------- windowed MHSA -----------------------------------------------------
__global__ __launch_bounds__(128)
void k_winattn(const bf16* __restrict__ qb, const bf16* __restrict__ kb,
               const bf16* __restrict__ vbT, const float* __restrict__ rel_bias,
               const float* __restrict__ mask, bf16* __restrict__ xo_pre) {
  __shared__ __attribute__((aligned(16))) bf16 P[64 * 80];
  const int bh = blockIdx.x;
  const int b = bh >> 3, h = bh & 7;
  const int lane = threadIdx.x & 31, wave = threadIdx.x >> 5;
  const int lh = lane >> 4, ll = lane & 15;
  const int m0 = wave * 16;
  const bf16* qp = qb + bh * N_PAD * HD;
  const bf16* kp = kb + bh * N_PAD * HD;
  const bf16* vp = vbT + bh * HD * N_PAD;

  // S = q @ k^T (K = hd = 32); A shared across the 4 column tiles -> RA hint
  v16bf aq = load_a_bf16(qp, HD, m0 + ll, 0, lh);
  v8f S[4];
  {
    const bf16* p0 = kp + ll * HD + lh * 16;
    v8f z0 = {}, z1 = {}, z2 = {}, z3 = {};
    S[0] = WMMA(aq, cat16(*(const v8bf*)(p0),        *(const v8bf*)(p0 + 8)),        z0, true);
    S[1] = WMMA(aq, cat16(*(const v8bf*)(p0 + 512),  *(const v8bf*)(p0 + 520)),      z1, true);
    S[2] = WMMA(aq, cat16(*(const v8bf*)(p0 + 1024), *(const v8bf*)(p0 + 1032)),     z2, true);
    S[3] = WMMA(aq, cat16(*(const v8bf*)(p0 + 1536), *(const v8bf*)(p0 + 1544)),     z3, false);
  }

  const int wm = b & (NW - 1);
  int ki_[4], kj_[4];
  #pragma unroll
  for (int ct = 0; ct < 4; ++ct) {
    const int nc = ct * 16 + ll;
    ki_[ct] = nc / 7; kj_[ct] = nc - ki_[ct] * 7;
  }
  #pragma unroll
  for (int r = 0; r < 8; ++r) {
    const int nq = m0 + r + lh * 8;
    const int qi = nq / 7, qj = nq - qi * 7;
    float sv[4];
    float mx = -1e30f;
    #pragma unroll
    for (int ct = 0; ct < 4; ++ct) {
      const int nc = ct * 16 + ll;
      float s = S[ct][r];
      if (nq < N_TOK && nc < N_TOK) {
        const int rel = (qi - ki_[ct] + 6) * 13 + (qj - kj_[ct] + 6);
        s += rel_bias[rel * H_NUM + h] + mask[(wm * N_TOK + nq) * N_TOK + nc];
      } else {
        s = -1e30f;
      }
      sv[ct] = s;
      mx = fmaxf(mx, s);
    }
    #pragma unroll
    for (int off = 1; off < 16; off <<= 1) mx = fmaxf(mx, __shfl_xor(mx, off));
    float sum = 0.f;
    #pragma unroll
    for (int ct = 0; ct < 4; ++ct) {
      float pv = (sv[ct] > -1e29f) ? __expf(sv[ct] - mx) : 0.f;
      sv[ct] = pv;
      sum += pv;
    }
    #pragma unroll
    for (int off = 1; off < 16; off <<= 1) sum += __shfl_xor(sum, off);
    const float inv = 1.f / sum;
    #pragma unroll
    for (int ct = 0; ct < 4; ++ct)
      P[nq * 80 + ct * 16 + ll] = (bf16)(sv[ct] * inv);
  }
  __syncthreads();

  // O = P @ v ; v^T d-major -> contiguous B fragments; A shared across 2 tiles
  v8f O[2] = {{}, {}};
  #pragma unroll
  for (int kt = 0; kt < 2; ++kt) {
    v16bf ap = load_a_bf16(P, 80, m0 + ll, kt * 32, lh);
    const bf16* p0 = vp + ll * N_PAD + kt * 32 + lh * 16;
    O[0] = WMMA(ap, cat16(*(const v8bf*)(p0),        *(const v8bf*)(p0 + 8)),    O[0], true);
    O[1] = WMMA(ap, cat16(*(const v8bf*)(p0 + 1024), *(const v8bf*)(p0 + 1032)), O[1], false);
  }
  #pragma unroll
  for (int ct = 0; ct < 2; ++ct) {
    #pragma unroll
    for (int r = 0; r < 8; ++r) {
      const int n = m0 + r + lh * 8;
      if (n < N_TOK)
        xo_pre[(b * N_TOK + n) * C_DIM + h * HD + ct * 16 + ll] = (bf16)O[ct][r];
    }
  }
}

// ---------------- proj GEMM (fp32 out + bf16 into t_x) ------------------------------
__global__ __launch_bounds__(128)
void k_proj(const bf16* __restrict__ A, const bf16* __restrict__ Wp,
            const float* __restrict__ bias, float* __restrict__ out,
            bf16* __restrict__ t_x) {
  __shared__ __attribute__((aligned(16))) bf16 At[16 * C_DIM];
  const int lane = threadIdx.x & 31, wave = threadIdx.x >> 5;
  const int lh = lane >> 4, ll = lane & 15;
  const int m0 = blockIdx.x * 16;
  const int nt0 = wave * 4;
  stage_a<C_DIM>(A, m0, At);
  v8f acc[4] = {{}, {}, {}, {}};
  mm_core4<C_DIM, 16>(At, Wp, nt0, lane, acc);
  #pragma unroll
  for (int j = 0; j < 4; ++j) {
    const int col = (nt0 + j) * 16 + ll;
    const float bv = bias[col];
    #pragma unroll
    for (int r = 0; r < 8; ++r) {
      int row = m0 + r + lh * 8;
      float v = acc[j][r] + bv;
      out[row * C_DIM + col] = v;
      t_x[row * TC_DIM + col] = (bf16)v;
    }
  }
}

// ---------------- t_x[:, 256:384] = [depth, seg] ------------------------------------
__global__ __launch_bounds__(256)
void k_fill_tokens(const bf16* __restrict__ db, const bf16* __restrict__ sb,
                   bf16* __restrict__ t_x) {
  int i = blockIdx.x * blockDim.x + threadIdx.x;
  if (i >= ROWS * CT_DIM) return;
  int row = i >> 6;
  int j = i & 63;
  t_x[row * TC_DIM + C_DIM + j]          = db[i];
  t_x[row * TC_DIM + C_DIM + CT_DIM + j] = sb[i];
}

// ---------------- token-q GEMM, transposed d-major output ---------------------------
__global__ __launch_bounds__(128)
void k_tok_q(const bf16* __restrict__ T, const bf16* __restrict__ Wp,
             const float* __restrict__ bias, bf16* __restrict__ outqT) {
  __shared__ __attribute__((aligned(16))) bf16 At[16 * CT_DIM];
  const int lane = threadIdx.x & 31, wave = threadIdx.x >> 5;
  const int lh = lane >> 4, ll = lane & 15;
  const int m0 = blockIdx.x * 16;
  stage_a<CT_DIM>(T, m0, At);
  v8f acc = mm_core1<CT_DIM, 4>(At, Wp, wave, lane);
  const int col = wave * 16 + ll;
  const int h = col >> 3, d = col & 7;
  const float bv = bias[col];
  int bi0, nr0; win_decomp(m0, bi0, nr0);
  #pragma unroll
  for (int r = 0; r < 8; ++r) {
    int t = nr0 + r + lh * 8;
    int bi = bi0 + (t >= N_TOK);
    int n  = t - ((t >= N_TOK) ? N_TOK : 0);
    outqT[((bi * H_NUM + h) * CTH + d) * N_PAD + n] = (bf16)((acc[r] + bv) * SCALE);
  }
}

// ---------------- t_k (t-major) / t_v (n-major) GEMM --------------------------------
__global__ __launch_bounds__(128)
void k_gkv(const bf16* __restrict__ A, const bf16* __restrict__ Wp,
           const float* __restrict__ bias, bf16* __restrict__ outp, int tmajor) {
  __shared__ __attribute__((aligned(16))) bf16 At[16 * TC_DIM];
  const int lane = threadIdx.x & 31, wave = threadIdx.x >> 5;
  const int lh = lane >> 4, ll = lane & 15;
  const int m0 = blockIdx.x * 16;
  const int nt0 = (blockIdx.y * 4 + wave) * 2;
  stage_a<TC_DIM>(A, m0, At);
  v8f acc[2] = {{}, {}};
  mm_core2<TC_DIM, 24>(At, Wp, nt0, lane, acc);
  int bi0, nr0; win_decomp(m0, bi0, nr0);
  #pragma unroll
  for (int j = 0; j < 2; ++j) {
    const int col = (nt0 + j) * 16 + ll;
    const int h = col / TH_DIM, t = col - h * TH_DIM;
    const float bv = bias[col];
    #pragma unroll
    for (int r = 0; r < 8; ++r) {
      int tt = nr0 + r + lh * 8;
      int bi = bi0 + (tt >= N_TOK);
      int n  = tt - ((tt >= N_TOK) ? N_TOK : 0);
      int bh = bi * H_NUM + h;
      float v = acc[j][r] + bv;
      if (tmajor)
        outp[(bh * TH_DIM + t) * N_PAD + n] = (bf16)v;   // t_k: (B,H,th,Npad)
      else
        outp[(bh * N_PAD + n) * TH_DIM + t] = (bf16)v;   // t_v: (B,H,Npad,th)
    }
  }
}

// ---------------- token cross-attention ---------------------------------------------
__global__ __launch_bounds__(128)
void k_tok_attn(const bf16* __restrict__ qT, const bf16* __restrict__ tkT,
                const bf16* __restrict__ tv, bf16* __restrict__ ob) {
  __shared__ __attribute__((aligned(16))) bf16 AL[4][16 * 64];
  const int lane = threadIdx.x & 31, wave = threadIdx.x >> 5;
  const int lh = lane >> 4, ll = lane & 15;
  const int bh = blockIdx.x * 4 + wave;
  const int b = bh >> 3, h = bh & 7;
  const bf16* qp = qT + bh * CTH * N_PAD;     // (cth, Npad) d-major
  const bf16* kp = tkT + bh * TH_DIM * N_PAD; // (th, Npad) t-major
  const bf16* vp = tv + bh * N_PAD * TH_DIM;  // (Npad, th) n-major
  bf16* al = AL[wave];

  // S[d,t] = sum_n q^T[d,n] k^T[t,n]; A shared across 3 tiles -> RA hints
  v8f S[3] = {{}, {}, {}};
  const int dd = ll & 7;
  #pragma unroll
  for (int kt = 0; kt < 2; ++kt) {
    v16bf a = load_a_bf16(qp, N_PAD, dd, kt * 32, lh);
    const bf16* p0 = kp + ll * N_PAD + kt * 32 + lh * 16;
    S[0] = WMMA(a, cat16(*(const v8bf*)(p0),        *(const v8bf*)(p0 + 8)),    S[0], true);
    S[1] = WMMA(a, cat16(*(const v8bf*)(p0 + 1024), *(const v8bf*)(p0 + 1032)), S[1], true);
    S[2] = WMMA(a, cat16(*(const v8bf*)(p0 + 2048), *(const v8bf*)(p0 + 2056)), S[2], false);
  }

  // softmax over t (48 valid)
  #pragma unroll
  for (int r = 0; r < 8; ++r) {
    const int rowd = r + lh * 8;
    float sv[3];
    float mx = -1e30f;
    #pragma unroll
    for (int ct = 0; ct < 3; ++ct) { sv[ct] = S[ct][r]; mx = fmaxf(mx, sv[ct]); }
    #pragma unroll
    for (int off = 1; off < 16; off <<= 1) mx = fmaxf(mx, __shfl_xor(mx, off));
    float sum = 0.f;
    #pragma unroll
    for (int ct = 0; ct < 3; ++ct) { sv[ct] = __expf(sv[ct] - mx); sum += sv[ct]; }
    #pragma unroll
    for (int off = 1; off < 16; off <<= 1) sum += __shfl_xor(sum, off);
    const float inv = 1.f / sum;
    #pragma unroll
    for (int ct = 0; ct < 3; ++ct)
      al[rowd * 64 + ct * 16 + ll] = (bf16)(sv[ct] * inv);
    al[rowd * 64 + 48 + ll] = (bf16)0.f;  // zero K-pad 48..63
  }
  __syncthreads();

  // O[d,n] = sum_t a[d,t] v[n,t]; K padded 48->64 (A zero in pad)
  v8f O[4] = {{}, {}, {}, {}};
  #pragma unroll
  for (int kt = 0; kt < 2; ++kt) {
    v16bf a = load_a_bf16(al, 64, ll, kt * 32, lh);
    const bf16* p0 = vp + ll * TH_DIM + kt * 32 + lh * 16;
    O[0] = WMMA(a, cat16(*(const v8bf*)(p0),        *(const v8bf*)(p0 + 8)),    O[0], true);
    O[1] = WMMA(a, cat16(*(const v8bf*)(p0 + 768),  *(const v8bf*)(p0 + 776)),  O[1], true);
    O[2] = WMMA(a, cat16(*(const v8bf*)(p0 + 1536), *(const v8bf*)(p0 + 1544)), O[2], true);
    O[3] = WMMA(a, cat16(*(const v8bf*)(p0 + 2304), *(const v8bf*)(p0 + 2312)), O[3], false);
  }
  if (lh == 0) {
    #pragma unroll
    for (int ct = 0; ct < 4; ++ct) {
      #pragma unroll
      for (int r = 0; r < 8; ++r) {
        const int n = ct * 16 + ll;
        if (n < N_TOK)
          ob[(b * N_TOK + n) * CT_DIM + h * CTH + r] = (bf16)O[ct][r];
      }
    }
  }
}

// ---------------- out = o @ pd_w + pd_b (fp32) --------------------------------------
__global__ __launch_bounds__(128)
void k_pd(const bf16* __restrict__ A, const bf16* __restrict__ Wp,
          const float* __restrict__ bias, float* __restrict__ out) {
  __shared__ __attribute__((aligned(16))) bf16 At[16 * CT_DIM];
  const int lane = threadIdx.x & 31, wave = threadIdx.x >> 5;
  const int lh = lane >> 4, ll = lane & 15;
  const int m0 = blockIdx.x * 16;
  stage_a<CT_DIM>(A, m0, At);
  v8f acc = mm_core1<CT_DIM, 4>(At, Wp, wave, lane);
  const int col = wave * 16 + ll;
  const float bv = bias[col];
  #pragma unroll
  for (int r = 0; r < 8; ++r) {
    int row = m0 + r + lh * 8;
    out[row * CT_DIM + col] = acc[r] + bv;
  }
}

extern "C" void kernel_launch(void* const* d_in, const int* in_sizes, int n_in,
                              void* d_out, int out_size, void* d_ws, size_t ws_size,
                              hipStream_t stream) {
  (void)in_sizes; (void)n_in; (void)out_size; (void)ws_size;
  const float* x       = (const float*)d_in[0];
  const float* mask    = (const float*)d_in[1];
  const float* depth   = (const float*)d_in[2];
  const float* seg     = (const float*)d_in[3];
  const float* qkv_w   = (const float*)d_in[4];
  const float* qkv_b   = (const float*)d_in[5];
  const float* rel_b   = (const float*)d_in[6];
  const float* proj_w  = (const float*)d_in[7];
  const float* proj_b  = (const float*)d_in[8];
  const float* dth_q_w = (const float*)d_in[9];
  const float* dth_q_b = (const float*)d_in[10];
  const float* seg_q_w = (const float*)d_in[11];
  const float* seg_q_b = (const float*)d_in[12];
  const float* gk_w    = (const float*)d_in[13];
  const float* gk_b    = (const float*)d_in[14];
  const float* gv_w    = (const float*)d_in[15];
  const float* gv_b    = (const float*)d_in[16];
  const float* pd_w    = (const float*)d_in[17];
  const float* pd_b    = (const float*)d_in[18];

  float* out_xo = (float*)d_out;
  float* out_d  = out_xo + (long)ROWS * C_DIM;
  float* out_s  = out_d + (long)ROWS * CT_DIM;

  bf16* w = (bf16*)d_ws;
  bf16* xb  = w;     w += (long)ROWS * C_DIM;
  bf16* db  = w;     w += (long)ROWS * CT_DIM;
  bf16* sb  = w;     w += (long)ROWS * CT_DIM;
  bf16* qb  = w;     w += (long)BH_NUM * N_PAD * HD;
  bf16* kb  = w;     w += (long)BH_NUM * N_PAD * HD;
  bf16* vbT = w;     w += (long)BH_NUM * HD * N_PAD;
  bf16* xo_pre = w;  w += (long)ROWS * C_DIM;
  bf16* t_x = w;     w += (long)ROWS * TC_DIM;
  bf16* dqT = w;     w += (long)BH_NUM * CTH * N_PAD;
  bf16* sqT = w;     w += (long)BH_NUM * CTH * N_PAD;
  bf16* tkT = w;     w += (long)BH_NUM * TH_DIM * N_PAD;
  bf16* tv  = w;     w += (long)BH_NUM * N_PAD * TH_DIM;
  bf16* od  = w;     w += (long)ROWS * CT_DIM;
  bf16* os_ = w;     w += (long)ROWS * CT_DIM;
  bf16* qkv_wp  = w; w += (long)C_DIM * 3 * C_DIM;
  bf16* proj_wp = w; w += (long)C_DIM * C_DIM;
  bf16* dth_wp  = w; w += (long)CT_DIM * CT_DIM;
  bf16* seg_wp  = w; w += (long)CT_DIM * CT_DIM;
  bf16* gk_wp   = w; w += (long)TC_DIM * TC_DIM;
  bf16* gv_wp   = w; w += (long)TC_DIM * TC_DIM;
  bf16* pd_wp   = w;

  // Padded n-rows participate in the n-contraction of token attention -> zero them.
  hipMemsetAsync(dqT, 0, (size_t)BH_NUM * CTH * N_PAD * sizeof(bf16), stream);
  hipMemsetAsync(sqT, 0, (size_t)BH_NUM * CTH * N_PAD * sizeof(bf16), stream);
  hipMemsetAsync(tkT, 0, (size_t)BH_NUM * TH_DIM * N_PAD * sizeof(bf16), stream);

  dim3 b256(256);
  k_cvt_bf16<<<dim3((ROWS * C_DIM / 4 + 255) / 256), b256, 0, stream>>>(x, xb, ROWS * C_DIM);
  k_cvt_bf16<<<dim3((ROWS * CT_DIM / 4 + 255) / 256), b256, 0, stream>>>(depth, db, ROWS * CT_DIM);
  k_cvt_bf16<<<dim3((ROWS * CT_DIM / 4 + 255) / 256), b256, 0, stream>>>(seg, sb, ROWS * CT_DIM);
  k_pack_w<<<dim3((C_DIM * 3 * C_DIM + 255) / 256), b256, 0, stream>>>(qkv_w, C_DIM, 3 * C_DIM, qkv_wp);
  k_pack_w<<<dim3((C_DIM * C_DIM + 255) / 256), b256, 0, stream>>>(proj_w, C_DIM, C_DIM, proj_wp);
  k_pack_w<<<dim3((CT_DIM * CT_DIM + 255) / 256), b256, 0, stream>>>(dth_q_w, CT_DIM, CT_DIM, dth_wp);
  k_pack_w<<<dim3((CT_DIM * CT_DIM + 255) / 256), b256, 0, stream>>>(seg_q_w, CT_DIM, CT_DIM, seg_wp);
  k_pack_w<<<dim3((TC_DIM * TC_DIM + 255) / 256), b256, 0, stream>>>(gk_w, TC_DIM, TC_DIM, gk_wp);
  k_pack_w<<<dim3((TC_DIM * TC_DIM + 255) / 256), b256, 0, stream>>>(gv_w, TC_DIM, TC_DIM, gv_wp);
  k_pack_w<<<dim3((CT_DIM * CT_DIM + 255) / 256), b256, 0, stream>>>(pd_w, CT_DIM, CT_DIM, pd_wp);

  dim3 blk(128);
  k_qkv<<<dim3(RT, 3), blk, 0, stream>>>(xb, qkv_wp, qkv_b, qb, kb, vbT);
  k_winattn<<<dim3(BH_NUM), blk, 0, stream>>>(qb, kb, vbT, rel_b, mask, xo_pre);
  k_proj<<<dim3(RT, 1), blk, 0, stream>>>(xo_pre, proj_wp, proj_b, out_xo, t_x);
  k_fill_tokens<<<dim3((ROWS * CT_DIM + 255) / 256), b256, 0, stream>>>(db, sb, t_x);
  k_tok_q<<<dim3(RT), blk, 0, stream>>>(db, dth_wp, dth_q_b, dqT);
  k_tok_q<<<dim3(RT), blk, 0, stream>>>(sb, seg_wp, seg_q_b, sqT);
  k_gkv<<<dim3(RT, 3), blk, 0, stream>>>(t_x, gk_wp, gk_b, tkT, 1);
  k_gkv<<<dim3(RT, 3), blk, 0, stream>>>(t_x, gv_wp, gv_b, tv, 0);
  k_tok_attn<<<dim3(BH_NUM / 4), blk, 0, stream>>>(dqT, tkT, tv, od);
  k_tok_attn<<<dim3(BH_NUM / 4), blk, 0, stream>>>(sqT, tkT, tv, os_);
  k_pd<<<dim3(RT), blk, 0, stream>>>(od, pd_wp, pd_b, out_d);
  k_pd<<<dim3(RT), blk, 0, stream>>>(os_, pd_wp, pd_b, out_s);
}
